// VAE_70360154243449
// MI455X (gfx1250) — matrix operand order repaired and checked by reading, not compiled, not run on hardware
//
#include <hip/hip_runtime.h>
#include <hip/hip_bf16.h>
#include <math.h>

#define B_SZ 4096
#define TTOT 40
#define TCTX 32

typedef __attribute__((ext_vector_type(16))) __bf16 v16bf;
typedef __attribute__((ext_vector_type(8)))  float  v8f;

__device__ __forceinline__ float eluf(float x){ return x > 0.f ? x : expm1f(x); }
__device__ __forceinline__ float softplusf(float x){ return x > 15.f ? x : log1pf(expf(x)); }

// ---------------- elementwise f32 -> bf16 ----------------
__global__ void k_cvt_bf16(const float* __restrict__ in, __bf16* __restrict__ out, int n){
  for (int i = blockIdx.x*blockDim.x + threadIdx.x; i < n; i += gridDim.x*blockDim.x)
    out[i] = (__bf16)in[i];
}

// pack W1[:, :512, :] (T,520,8) -> row-major (512, T*8) bf16
__global__ void k_pack_w1(const float* __restrict__ W1, __bf16* __restrict__ out){
  int i = blockIdx.x*blockDim.x + threadIdx.x;
  const int n = 512 * TTOT * 8;
  if (i >= n) return;
  int c = i % (TTOT*8); int d = i / (TTOT*8);
  int t = c >> 3, j = c & 7;
  out[i] = (__bf16)W1[(size_t)t*520*8 + (size_t)d*8 + j];
}

// W_eff[(t*64+p), n] = sum_d W1[p,d] * W2[t*512+d, n]   (f32 compute, bf16 out, row-major)
__global__ void k_weff(const float* __restrict__ W1, const float* __restrict__ W2,
                       __bf16* __restrict__ out, int T){
  int i = blockIdx.x*blockDim.x + threadIdx.x;
  int tot = T*64*512;
  if (i >= tot) return;
  int n = i % 512; int row = i / 512;
  int p = row % 64, t = row / 64;
  const float* w1 = W1 + (size_t)p*512;
  const float* w2 = W2 + (size_t)t*512*512 + n;
  float s = 0.f;
  #pragma unroll 8
  for (int d = 0; d < 512; ++d) s += w1[d] * w2[(size_t)d*512];
  out[i] = (__bf16)s;
}

// LDS tile transpose: src (R x C) -> dst (C x R), bf16, R%32==0, C%32==0
__global__ void k_transpose_bf16(const __bf16* __restrict__ src, __bf16* __restrict__ dst,
                                 int R, int C){
  __shared__ __bf16 tile[32][33];
  int c0 = blockIdx.x * 32, r0 = blockIdx.y * 32;
  int tx = threadIdx.x, ty = threadIdx.y;   // block (32, 8)
  for (int i = ty; i < 32; i += 8)
    tile[i][tx] = src[(size_t)(r0 + i)*C + c0 + tx];
  __syncthreads();
  for (int i = ty; i < 32; i += 8)
    dst[(size_t)(c0 + i)*R + r0 + tx] = tile[tx][i];
}

// ---------------- bias_eff, parallel 3-stage ----------------
// pev[t*512+d] = b1[d] + PE(t,d)
__global__ void k_pe(const float* __restrict__ b1, float* __restrict__ pev, int T){
  int i = blockIdx.x*blockDim.x + threadIdx.x;
  if (i >= T*512) return;
  int t = i / 512, d = i % 512;
  float ex  = (float)(d & ~1) / 512.0f;
  float ang = (float)t * powf(10000.0f, -ex);
  pev[i] = b1[d] + ((d & 1) ? cosf(ang) : sinf(ang));
}
// part[t*512+n] = sum_d pev[t*512+d] * W2[(t*512+d)*512 + n]
__global__ void k_biaspart(const float* __restrict__ pev, const float* __restrict__ W2,
                           float* __restrict__ part, int T){
  int i = blockIdx.x*blockDim.x + threadIdx.x;
  if (i >= T*512) return;
  int n = i % 512, t = i / 512;
  const float* w2 = W2 + (size_t)t*512*512 + n;
  const float* pv = pev + (size_t)t*512;
  float s = 0.f;
  #pragma unroll 8
  for (int d = 0; d < 512; ++d) s += pv[d] * w2[(size_t)d*512];
  part[i] = s;
}
// out[n] = b2[n] + sum_t part[t*512+n]
__global__ void k_biasred(const float* __restrict__ part, const float* __restrict__ b2,
                          float* __restrict__ out, int T){
  int n = blockIdx.x*blockDim.x + threadIdx.x;
  if (n >= 512) return;
  float s = b2[n];
  for (int t = 0; t < T; ++t) s += part[(size_t)t*512 + n];
  out[n] = s;
}

// ---------------- bf16 WMMA GEMM, double-buffered LDS ----------------
// C[M,N] = A[M,K] @ B[K,N] + bias[N]
// A row-major (M x K) bf16; BT is B transposed: (N x K) bf16.  M%128==0, N%64==0, K%32==0
template<bool STORE_BF16>
__global__ __launch_bounds__(256) void k_gemm(const __bf16* __restrict__ A,
                                              const __bf16* __restrict__ BT,
                                              const float* __restrict__ bias,
                                              void* __restrict__ Cout,
                                              int M, int N, int K){
  __shared__ __bf16 As[2][128][40];   // stride 40 bf16 = 80B: 16B-aligned, conflict-free
  __shared__ __bf16 Bs[2][64][40];
  const int tid  = threadIdx.x;
  const int lane = tid & 31;
  const int w    = tid >> 5;
  const int wm   = w & 3, wn = w >> 2;       // 4x2 wave grid, 32x32 out per wave
  const int m16  = lane & 15, half = lane >> 4;
  const int m0   = blockIdx.y * 128;
  const int n0   = blockIdx.x * 64;

  const v8f vzero = {0.f,0.f,0.f,0.f,0.f,0.f,0.f,0.f};
  v8f acc[2][2];
  #pragma unroll
  for (int i = 0; i < 2; ++i)
    #pragma unroll
    for (int j = 0; j < 2; ++j) acc[i][j] = vzero;

  const int ar = tid >> 1, ac = (tid & 1) * 16;   // A: 128x32 tile, 16 bf16/thread
  const int br = tid >> 2, bc = (tid & 3) * 8;    // B: 64x32 tile,   8 bf16/thread
  const __bf16* Ap = A  + (size_t)(m0 + ar)*K + ac;
  const __bf16* Bp = BT + (size_t)(n0 + br)*K + bc;

  { // prologue: tile 0
    uint4 a0 = *(const uint4*)Ap;
    uint4 a1 = *(const uint4*)(Ap + 8);
    uint4 b0 = *(const uint4*)Bp;
    *(uint4*)(&As[0][ar][ac])     = a0;
    *(uint4*)(&As[0][ar][ac + 8]) = a1;
    *(uint4*)(&Bs[0][br][bc])     = b0;
  }
  __syncthreads();

  const int nk = K >> 5;
  for (int it = 0; it < nk; ++it){
    const int cur = it & 1, nxt = cur ^ 1;
    uint4 na0 = {0,0,0,0}, na1 = {0,0,0,0}, nb0 = {0,0,0,0};
    if (it + 1 < nk){
      const __bf16* ap = Ap + ((size_t)(it + 1) << 5);
      const __bf16* bp = Bp + ((size_t)(it + 1) << 5);
      na0 = *(const uint4*)ap;
      na1 = *(const uint4*)(ap + 8);
      nb0 = *(const uint4*)bp;
      if (it + 2 < nk){
        __builtin_prefetch(ap + 32, 0, 0);   // global_prefetch two tiles ahead
        __builtin_prefetch(bp + 32, 0, 0);
      }
    }

    // A frag: lane m = lane&15; elems j<8 -> k=8*half+j ; j>=8 -> k=16+8*half+(j-8)
    v16bf af[2], bfr[2];
    #pragma unroll
    for (int i = 0; i < 2; ++i){
      const __bf16* base = &As[cur][wm*32 + i*16 + m16][0];
      ((uint4*)&af[i])[0] = *(const uint4*)(base + 8*half);
      ((uint4*)&af[i])[1] = *(const uint4*)(base + 16 + 8*half);
    }
    // B frag: lane n = lane&15; elems j<8 -> k=8*half+j ; j>=8 -> k=16+8*half+(j-8)
    #pragma unroll
    for (int j = 0; j < 2; ++j){
      const __bf16* base = &Bs[cur][wn*32 + j*16 + m16][0];
      ((uint4*)&bfr[j])[0] = *(const uint4*)(base + 8*half);
      ((uint4*)&bfr[j])[1] = *(const uint4*)(base + 16 + 8*half);
    }
    #pragma unroll
    for (int i = 0; i < 2; ++i)
      #pragma unroll
      for (int j = 0; j < 2; ++j)
        acc[i][j] = __builtin_amdgcn_wmma_f32_16x16x32_bf16(
            false, af[i], false, bfr[j], (short)0, acc[i][j], false, false);

    if (it + 1 < nk){
      *(uint4*)(&As[nxt][ar][ac])     = na0;
      *(uint4*)(&As[nxt][ar][ac + 8]) = na1;
      *(uint4*)(&Bs[nxt][br][bc])     = nb0;
    }
    __syncthreads();
  }

  // C/D layout: VGPR e -> M = e + 8*half, N = lane&15
  #pragma unroll
  for (int i = 0; i < 2; ++i)
    #pragma unroll
    for (int j = 0; j < 2; ++j){
      int n = n0 + wn*32 + j*16 + m16;
      float bvv = bias ? bias[n] : 0.f;
      int mb = m0 + wm*32 + i*16 + 8*half;
      #pragma unroll
      for (int e = 0; e < 8; ++e){
        float v = acc[i][j][e] + bvv;
        size_t idx = (size_t)(mb + e)*N + n;
        if (STORE_BF16) ((__bf16*)Cout)[idx] = (__bf16)v;
        else            ((float*)Cout)[idx]  = v;
      }
    }
}

// ---------------- sequential latent scans (1 thread per batch element) ----------------
__global__ void k_prior_scan(const float* __restrict__ pre, const float* __restrict__ W1,
                             const float* __restrict__ W2, const float* __restrict__ b2,
                             const float* __restrict__ eps,
                             float* __restrict__ oz, float* __restrict__ om,
                             float* __restrict__ olv){
  int b = blockIdx.x*blockDim.x + threadIdx.x;
  if (b >= B_SZ) return;
  float zp[8];
  #pragma unroll
  for (int j = 0; j < 8; ++j) zp[j] = 0.f;
  for (int t = 0; t < TTOT; ++t){
    float a[8];
    const float* pr = pre + (size_t)b*(TTOT*8) + t*8;      // includes b1 (GEMM bias)
    #pragma unroll
    for (int j = 0; j < 8; ++j) a[j] = pr[j];
    const float* wz = W1 + (size_t)t*4160 + 4096;          // rows 512..519
    #pragma unroll
    for (int i = 0; i < 8; ++i)
      #pragma unroll
      for (int j = 0; j < 8; ++j) a[j] += zp[i]*wz[i*8+j];
    float h[8];
    #pragma unroll
    for (int j = 0; j < 8; ++j) h[j] = eluf(a[j]);
    const float* w2 = W2 + (size_t)t*64;
    const float* bb = b2 + t*8;
    const float* ev = eps + ((size_t)t*B_SZ + b)*8;
    size_t ob = ((size_t)t*B_SZ + b)*8;
    #pragma unroll
    for (int k = 0; k < 8; ++k){
      float m = bb[k];
      #pragma unroll
      for (int j = 0; j < 8; ++j) m += h[j]*w2[j*8+k];
      float z = m + ev[k];                                  // sqrt(PRIOR_VAR)=1
      oz[ob+k] = z; om[ob+k] = m; olv[ob+k] = 0.f;          // log(1.0)=0
      zp[k] = 0.1f*zp[k] + 0.9f*z;
    }
  }
}

__global__ void k_post_scan(const float* __restrict__ pre, const float* __restrict__ W1,
                            const float* __restrict__ W2, const float* __restrict__ b2,
                            const float* __restrict__ eps,
                            float* __restrict__ oz, float* __restrict__ om,
                            float* __restrict__ olv){
  int b = blockIdx.x*blockDim.x + threadIdx.x;
  if (b >= B_SZ) return;
  float zp[8];
  #pragma unroll
  for (int j = 0; j < 8; ++j) zp[j] = 0.f;
  for (int t = 0; t < TTOT; ++t){
    float a[8];
    const float* pr = pre + (size_t)b*(TTOT*8) + t*8;
    #pragma unroll
    for (int j = 0; j < 8; ++j) a[j] = pr[j];
    const float* wz = W1 + (size_t)t*4160 + 4096;
    #pragma unroll
    for (int i = 0; i < 8; ++i)
      #pragma unroll
      for (int j = 0; j < 8; ++j) a[j] += zp[i]*wz[i*8+j];
    float h[8];
    #pragma unroll
    for (int j = 0; j < 8; ++j) h[j] = eluf(a[j]);
    const float* w2 = W2 + (size_t)t*128;                  // (8,16)
    const float* bb = b2 + t*16;
    const float* ev = eps + ((size_t)t*B_SZ + b)*8;
    size_t ob = ((size_t)t*B_SZ + b)*8;
    #pragma unroll
    for (int k = 0; k < 8; ++k){
      float m  = bb[k];
      float lv = bb[8+k];
      #pragma unroll
      for (int j = 0; j < 8; ++j){ m += h[j]*w2[j*16+k]; lv += h[j]*w2[j*16+8+k]; }
      float z = m + expf(0.5f*lv)*ev[k];
      oz[ob+k] = z; om[ob+k] = m; olv[ob+k] = lv;
      zp[k] = 0.1f*zp[k] + 0.9f*z;
    }
  }
}

// ---------------- spline head: ~0.5 GB output, bandwidth-bound ----------------
__global__ __launch_bounds__(256) void k_spline(const float* __restrict__ post_z,
    const float* __restrict__ sW1, const float* __restrict__ sb1,
    const float* __restrict__ sW2, const float* __restrict__ sb2,
    float* __restrict__ gamma, float* __restrict__ beta){
  __shared__ float W2s[8*768];
  __shared__ float b2s[768];
  __shared__ float h1s[16][8];
  const int blocksPerT = B_SZ/16;
  int t  = blockIdx.x / blocksPerT;
  int b0 = (blockIdx.x % blocksPerT) * 16;
  int tid = threadIdx.x;
  const float* w2 = sW2 + (size_t)t*8*768;
  for (int i = tid; i < 8*768; i += 256) W2s[i] = w2[i];
  for (int i = tid; i < 768;  i += 256) b2s[i] = sb2[(size_t)t*768 + i];
  if (tid < 128){
    int bi = tid >> 3, j = tid & 7;
    const float* z = post_z + ((size_t)t*B_SZ + b0 + bi)*8;
    float s = sb1[t*8 + j];
    #pragma unroll
    for (int i2 = 0; i2 < 8; ++i2) s += z[i2]*sW1[(size_t)t*64 + i2*8 + j];
    h1s[bi][j] = eluf(s);
  }
  __syncthreads();
  for (int idx = tid; idx < 16*768; idx += 256){
    int bi = idx / 768, o = idx % 768;
    float v = b2s[o];
    #pragma unroll
    for (int j = 0; j < 8; ++j) v += h1s[bi][j]*W2s[j*768 + o];
    int p = o / 12, m = o % 12;
    size_t row = (size_t)t*B_SZ + b0 + bi;
    if (m == 0) gamma[row*64 + p] = v;
    else        beta[(row*64 + p)*11 + (m-1)] = softplusf(v);
  }
}

__global__ void k_delta(float* __restrict__ out){
  int i = threadIdx.x;
  if (i < 11) out[i] = (float)i * 0.1f;
}

// ---------------- launcher ----------------
extern "C" void kernel_launch(void* const* d_in, const int* in_sizes, int n_in,
                              void* d_out, int out_size, void* d_ws, size_t ws_size,
                              hipStream_t stream){
  (void)in_sizes; (void)n_in; (void)out_size; (void)ws_size;
  const float* ctx  = (const float*)d_in[0];
  const float* tgt  = (const float*)d_in[1];
  const float* W_C1 = (const float*)d_in[2];
  const float* b_C1 = (const float*)d_in[3];
  const float* W_C2 = (const float*)d_in[4];
  const float* b_C2 = (const float*)d_in[5];
  const float* W_T1 = (const float*)d_in[6];
  const float* b_T1 = (const float*)d_in[7];
  const float* W_T2 = (const float*)d_in[8];
  const float* b_T2 = (const float*)d_in[9];
  const float* prW1 = (const float*)d_in[10];
  const float* prb1 = (const float*)d_in[11];
  const float* prW2 = (const float*)d_in[12];
  const float* prb2 = (const float*)d_in[13];
  const float* poW1 = (const float*)d_in[14];
  const float* pob1 = (const float*)d_in[15];
  const float* poW2 = (const float*)d_in[16];
  const float* pob2 = (const float*)d_in[17];
  const float* sW1  = (const float*)d_in[18];
  const float* sb1  = (const float*)d_in[19];
  const float* sW2  = (const float*)d_in[20];
  const float* sb2  = (const float*)d_in[21];
  const float* epr  = (const float*)d_in[22];
  const float* epo  = (const float*)d_in[23];

  char* wp = (char*)d_ws;
  auto walloc = [&](size_t bytes)->void*{ void* p = wp; wp += (bytes + 255) & ~(size_t)255; return p; };
  __bf16* ctx_b   = (__bf16*)walloc((size_t)4096*2048*2);
  __bf16* tgt_b   = (__bf16*)walloc((size_t)4096*2560*2);
  __bf16* weffC   = (__bf16*)walloc((size_t)2048*512*2);   // row-major (K x N)
  __bf16* weffT   = (__bf16*)walloc((size_t)2560*512*2);
  __bf16* weffC_T = (__bf16*)walloc((size_t)512*2048*2);   // transposed (N x K)
  __bf16* weffT_T = (__bf16*)walloc((size_t)512*2560*2);
  float*  biasC   = (float*)walloc(512*4);
  float*  biasT   = (float*)walloc(512*4);
  float*  peC     = (float*)walloc((size_t)TCTX*512*4);
  float*  peT     = (float*)walloc((size_t)TTOT*512*4);
  float*  partC   = (float*)walloc((size_t)TCTX*512*4);
  float*  partT   = (float*)walloc((size_t)TTOT*512*4);
  __bf16* hC2     = (__bf16*)walloc((size_t)4096*512*2);
  __bf16* hT2     = (__bf16*)walloc((size_t)4096*512*2);
  __bf16* wpPr    = (__bf16*)walloc((size_t)512*320*2);
  __bf16* wpPo    = (__bf16*)walloc((size_t)512*320*2);
  __bf16* wpPr_T  = (__bf16*)walloc((size_t)320*512*2);
  __bf16* wpPo_T  = (__bf16*)walloc((size_t)320*512*2);
  float*  prePr   = (float*)walloc((size_t)4096*320*4);
  float*  prePo   = (float*)walloc((size_t)4096*320*4);

  float* out = (float*)d_out;
  const size_t TB8 = (size_t)TTOT*B_SZ*8;
  float* o_pz    = out;
  float* o_pm    = out + TB8;
  float* o_plv   = out + 2*TB8;
  float* o_qz    = out + 3*TB8;
  float* o_qm    = out + 4*TB8;
  float* o_qlv   = out + 5*TB8;
  float* o_gamma = out + 6*TB8;
  float* o_beta  = o_gamma + (size_t)TTOT*B_SZ*64;
  float* o_delta = o_beta  + (size_t)TTOT*B_SZ*64*11;

  // --- input conversions / weight prep ---
  k_cvt_bf16<<<1024, 256, 0, stream>>>(ctx, ctx_b, 4096*2048);
  k_cvt_bf16<<<1024, 256, 0, stream>>>(tgt, tgt_b, 4096*2560);
  k_pack_w1<<<(512*320 + 255)/256, 256, 0, stream>>>(prW1, wpPr);
  k_pack_w1<<<(512*320 + 255)/256, 256, 0, stream>>>(poW1, wpPo);
  k_weff<<<(TCTX*64*512 + 255)/256, 256, 0, stream>>>(W_C1, W_C2, weffC, TCTX);
  k_weff<<<(TTOT*64*512 + 255)/256, 256, 0, stream>>>(W_T1, W_T2, weffT, TTOT);

  dim3 tb(32, 8);
  k_transpose_bf16<<<dim3(512/32, 2048/32), tb, 0, stream>>>(weffC, weffC_T, 2048, 512);
  k_transpose_bf16<<<dim3(512/32, 2560/32), tb, 0, stream>>>(weffT, weffT_T, 2560, 512);
  k_transpose_bf16<<<dim3(320/32,  512/32), tb, 0, stream>>>(wpPr,  wpPr_T,   512, 320);
  k_transpose_bf16<<<dim3(320/32,  512/32), tb, 0, stream>>>(wpPo,  wpPo_T,   512, 320);

  k_pe<<<(TCTX*512 + 255)/256, 256, 0, stream>>>(b_C1, peC, TCTX);
  k_pe<<<(TTOT*512 + 255)/256, 256, 0, stream>>>(b_T1, peT, TTOT);
  k_biaspart<<<(TCTX*512 + 255)/256, 256, 0, stream>>>(peC, W_C2, partC, TCTX);
  k_biaspart<<<(TTOT*512 + 255)/256, 256, 0, stream>>>(peT, W_T2, partT, TTOT);
  k_biasred<<<2, 256, 0, stream>>>(partC, b_C2, biasC, TCTX);
  k_biasred<<<2, 256, 0, stream>>>(partT, b_T2, biasT, TTOT);

  // --- WMMA GEMMs ---
  dim3 gEnc(512/64, 4096/128);
  k_gemm<true ><<<gEnc, 256, 0, stream>>>(ctx_b, weffC_T, biasC, hC2, 4096, 512, 2048);
  k_gemm<true ><<<gEnc, 256, 0, stream>>>(tgt_b, weffT_T, biasT, hT2, 4096, 512, 2560);
  dim3 gPre(320/64, 4096/128);
  k_gemm<false><<<gPre, 256, 0, stream>>>(hC2, wpPr_T, prb1, prePr, 4096, 320, 512);
  k_gemm<false><<<gPre, 256, 0, stream>>>(hT2, wpPo_T, pob1, prePo, 4096, 320, 512);

  // --- sequential scans + heads ---
  k_prior_scan<<<B_SZ/256, 256, 0, stream>>>(prePr, prW1, prW2, prb2, epr, o_pz, o_pm, o_plv);
  k_post_scan <<<B_SZ/256, 256, 0, stream>>>(prePo, poW1, poW2, pob2, epo, o_qz, o_qm, o_qlv);

  k_spline<<<TTOT*(B_SZ/16), 256, 0, stream>>>(o_qz, sW1, sb1, sW2, sb2, o_gamma, o_beta);
  k_delta<<<1, 32, 0, stream>>>(o_delta);
}
